// LinearAttention_61856118997263
// MI455X (gfx1250) — matrix operand order repaired and checked by reading, not compiled, not run on hardware
//
#include <hip/hip_runtime.h>
#include <hip/hip_bf16.h>
#include <math.h>
#include <stdint.h>

// ---------------- types ----------------
typedef __attribute__((ext_vector_type(16))) _Float16 v16h;
typedef __attribute__((ext_vector_type(8)))  _Float16 v8h;
typedef __attribute__((ext_vector_type(8)))  float    v8f;

#define NB    16          // batch
#define CH    128         // dim
#define NPIX  9216        // 96*96
#define HEADS 4
#define DH    32
#define SQRTD 11.313708498984761f    // sqrt(128)
#define QSCALE 0.17677669529663687f  // 32^-0.5

// ---------------- WMMA helpers ----------------
__device__ __forceinline__ v8f wmma_f16(v16h a, v16h b, v8f c) {
    return __builtin_amdgcn_wmma_f32_16x16x32_f16(
        false, a, false, b, (short)0, c, false, false);
}

// A fragment: row-major f16, per-lane row pointer, K-offset k0.
// lanes 0-15: K = k0+{0..15}; lanes 16-31: K = k0+{16..31}
__device__ __forceinline__ v16h load_a_frag(const _Float16* row, int k0, int lane) {
    int off = k0 + ((lane & 16) ? 16 : 0);
    v8h lo = *(const v8h*)(row + off);
    v8h hi = *(const v8h*)(row + off + 8);
    v16h f;
#pragma unroll
    for (int i = 0; i < 8; ++i) { f[i] = lo[i]; f[i + 8] = hi[i]; }
    return f;
}

// B fragment: per-lane column data contiguous along K.
// lanes 0-15: K = k0+{0..15}; lanes 16-31: K = k0+{16..31}
__device__ __forceinline__ v16h load_b_frag(const _Float16* col, int k0, int lane) {
    int off = k0 + ((lane & 16) ? 16 : 0);
    v8h a = *(const v8h*)(col + off);
    v8h b = *(const v8h*)(col + off + 8);
    v16h f;
#pragma unroll
    for (int i = 0; i < 8; ++i) { f[i] = a[i]; f[i + 8] = b[i]; }
    return f;
}

__device__ __forceinline__ v8f zero_v8f() {
    v8f z;
#pragma unroll
    for (int i = 0; i < 8; ++i) z[i] = 0.f;
    return z;
}

// ---------------- async global->LDS DMA (16B per lane, ASYNCcnt) ----------------
__device__ __forceinline__ void async_copy16(void* lds_dst, const void* gsrc) {
    unsigned l = (unsigned)(size_t)lds_dst;              // low 32 bits = LDS offset
    unsigned long long g = (unsigned long long)(size_t)gsrc;
    asm volatile("global_load_async_to_lds_b128 %0, %1, off"
                 :: "v"(l), "v"(g) : "memory");
}
__device__ __forceinline__ void async_wait0() {
    asm volatile("s_wait_asynccnt 0x0" ::: "memory");
}

// ---------------- K0: weights fp32 -> f16 ----------------
__global__ void __launch_bounds__(256) k_weights(const float* __restrict__ qkv_w,
                                                 const float* __restrict__ out_w,
                                                 _Float16* __restrict__ Wq,
                                                 _Float16* __restrict__ Wo) {
    int i = blockIdx.x * 256 + threadIdx.x;
    if (i < 384 * 128) Wq[i] = (_Float16)qkv_w[i];
    if (i < 128 * 128) Wo[i] = (_Float16)out_w[i];
}

// ---------------- K1: rmsnorm + qkv GEMM ----------------
// block: 64 pixels of one batch; 256 threads = 8 waves.
// qkv layout: [b][row 0..383][n] f16, rows 0-127=q, 128-255=k, 256-383=v
__global__ void __launch_bounds__(256) k_rmsnorm_qkv(
    const float* __restrict__ x, const float* __restrict__ norm_g,
    const _Float16* __restrict__ Wq, _Float16* __restrict__ qkv) {

    const int b  = blockIdx.y;
    const int n0 = blockIdx.x * 64;
    const int t  = threadIdx.x;

    __shared__ float    xraw[128][68];
    __shared__ float    psum[4][64];
    __shared__ float    sc[64];
    __shared__ _Float16 xnh[64][136];   // pixel-major normalized f16 (B source)

    const float* xb = x + (unsigned)b * (CH * NPIX) + n0;
    // stage x tile (128 ch x 64 px fp32) via async DMA: 2048 16B chunks
#pragma unroll
    for (int i = 0; i < 8; ++i) {
        int q = t + 256 * i;
        int r = q >> 4, c0 = (q & 15) * 4;               // 16 chunks per 64-float row
        async_copy16(&xraw[r][c0], xb + (unsigned)r * NPIX + c0);
    }
    async_wait0();
    __syncthreads();

    const int px = t & 63, qr = t >> 6;
    float part = 0.f;
#pragma unroll
    for (int i = 0; i < 32; ++i) {
        int c = qr * 32 + i;
        float v = xraw[c][px];
        part += v * v;
    }
    psum[qr][px] = part;
    __syncthreads();
    if (t < 64) {
        float s = psum[0][t] + psum[1][t] + psum[2][t] + psum[3][t];
        sc[t] = SQRTD / fmaxf(sqrtf(s), 1e-12f);
    }
    __syncthreads();
#pragma unroll
    for (int i = 0; i < 32; ++i) {
        int c = qr * 32 + i;
        xnh[px][c] = (_Float16)(xraw[c][px] * sc[px] * norm_g[c]);
    }
    __syncthreads();

    const int wave = t >> 5, lane = t & 31;
    for (int mi = 0; mi < 3; ++mi) {
        int mt = wave * 3 + mi;                          // M-tile of 24
        const _Float16* wrow = Wq + (unsigned)(mt * 16 + (lane & 15)) * 128;
        v16h afr[4];
#pragma unroll
        for (int ks = 0; ks < 4; ++ks) afr[ks] = load_a_frag(wrow, ks * 32, lane);

        for (int nt = 0; nt < 4; ++nt) {
            v8f acc = zero_v8f();
            const _Float16* col = &xnh[nt * 16 + (lane & 15)][0];
#pragma unroll
            for (int ks = 0; ks < 4; ++ks)
                acc = wmma_f16(afr[ks], load_b_frag(col, ks * 32, lane), acc);

            unsigned colg = (unsigned)(n0 + nt * 16 + (lane & 15));
            unsigned row0 = (unsigned)(b * 384 + mt * 16 + ((lane & 16) ? 8 : 0));
            unsigned idx  = row0 * NPIX + colg;
#pragma unroll
            for (int r = 0; r < 8; ++r)
                qkv[idx + (unsigned)r * NPIX] = (_Float16)acc[r];
        }
    }
}

// ---------------- K3: k softmax stats (max + sumexp over n + mem tokens) ----------------
__global__ void __launch_bounds__(256) k_kstats(const _Float16* __restrict__ qkv,
                                                const float* __restrict__ mem_kv,
                                                float* __restrict__ kmax,
                                                float* __restrict__ ksum) {
    const int row = blockIdx.x;        // b*128 + (h*32+d)
    const int rr  = row & 127;
    const int b   = row >> 7;
    const _Float16* kr = qkv + (unsigned)(b * 384 + 128 + rr) * NPIX;
    __shared__ float red[256];
    const int t = threadIdx.x;

    float m = -1e30f;
    for (int n = t; n < NPIX; n += 256) m = fmaxf(m, (float)kr[n]);
    if (t < 4) m = fmaxf(m, mem_kv[rr * 4 + t]);   // mem_kv[0][h][d][t]
    red[t] = m; __syncthreads();
    for (int s = 128; s > 0; s >>= 1) { if (t < s) red[t] = fmaxf(red[t], red[t + s]); __syncthreads(); }
    float mx = red[0];
    __syncthreads();

    float sum = 0.f;
    for (int n = t; n < NPIX; n += 256) sum += __expf((float)kr[n] - mx);
    if (t < 4) sum += __expf(mem_kv[rr * 4 + t] - mx);
    red[t] = sum; __syncthreads();
    for (int s = 128; s > 0; s >>= 1) { if (t < s) red[t] += red[t + s]; __syncthreads(); }
    if (t == 0) { kmax[row] = mx; ksum[row] = red[0]; }
}

// ---------------- K4: context[d][e] = softmax(k) . v^T ----------------
// one block (4 waves) per (b,h); wave -> 16x16 tile of the 32x32 output
__global__ void __launch_bounds__(128) k_context(const _Float16* __restrict__ qkv,
                                                 const float* __restrict__ mem_kv,
                                                 const float* __restrict__ kmax,
                                                 const float* __restrict__ ksum,
                                                 float* __restrict__ ctx) {
    const int bh = blockIdx.x;
    const int b = bh >> 2, h = bh & 3;
    const int wave = threadIdx.x >> 5, lane = threadIdx.x & 31;
    const int dHalf = wave >> 1, eHalf = wave & 1;

    const int dl = dHalf * 16 + (lane & 15);
    const int el = eHalf * 16 + (lane & 15);
    const _Float16* ka = qkv + (unsigned)(b * 384 + 128 + h * 32 + dl) * NPIX;
    const _Float16* vb = qkv + (unsigned)(b * 384 + 256 + h * 32 + el) * NPIX;
    const float mx  = kmax[bh * 32 + dl];
    const float inv = 1.f / ksum[bh * 32 + dl];

    v8f acc = zero_v8f();
    for (int n0 = 0; n0 < NPIX; n0 += 32) {
        int off = n0 + ((lane & 16) ? 16 : 0);
        v8h a0 = *(const v8h*)(ka + off);
        v8h a1 = *(const v8h*)(ka + off + 8);
        v16h af;
#pragma unroll
        for (int i = 0; i < 8; ++i) {
            af[i]     = (_Float16)(__expf((float)a0[i] - mx) * inv);
            af[i + 8] = (_Float16)(__expf((float)a1[i] - mx) * inv);
        }
        acc = wmma_f16(af, load_b_frag(vb, n0, lane), acc);
    }
    // memory-kv tail + store
#pragma unroll
    for (int r = 0; r < 8; ++r) {
        int d = dHalf * 16 + r + ((lane & 16) ? 8 : 0);
        float mxr  = kmax[bh * 32 + d];
        float invr = 1.f / ksum[bh * 32 + d];
        float add = 0.f;
#pragma unroll
        for (int j = 0; j < 4; ++j)
            add += __expf(mem_kv[(h * 32 + d) * 4 + j] - mxr) * invr *
                   mem_kv[512 + (h * 32 + el) * 4 + j];
        ctx[(unsigned)(bh * 32 + d) * 32 + el] = acc[r] + add;
    }
}

// ---------------- K5: q softmax (over d) + out = context^T . q ----------------
// block: (b,h) x 256 pixels; 4 waves. attn written into the (dead) k section: [b][n][128] f16.
__global__ void __launch_bounds__(128) k_qattn(_Float16* __restrict__ qkv,
                                               const float* __restrict__ ctx) {
    const int bh = blockIdx.y;
    const int b = bh >> 2, h = bh & 3;
    const int n0 = blockIdx.x * 256;
    const int t = threadIdx.x, wave = t >> 5, lane = t & 31;

    __shared__ _Float16 qs[32][264];   // staged q logits [d][px]
    __shared__ _Float16 qt[256][32];   // softmaxed, transposed [px][d]

    const _Float16* qb = qkv + (unsigned)(b * 384 + h * 32) * NPIX + n0;
    _Float16* attn = qkv + (unsigned)(b * 384 + 128) * NPIX;   // reuse k section

    // stage q tile (32 x 256 f16) via async DMA: 1024 16B chunks
#pragma unroll
    for (int i = 0; i < 8; ++i) {
        int q = t + 128 * i;
        int r = q >> 5, c0 = (q & 31) * 8;               // 32 chunks per 256-half row
        async_copy16(&qs[r][c0], qb + (unsigned)r * NPIX + c0);
    }
    async_wait0();
    __syncthreads();

    for (int p = t; p < 256; p += 128) {
        float vv[32], m = -1e30f;
#pragma unroll
        for (int d = 0; d < 32; ++d) { vv[d] = (float)qs[d][p]; m = fmaxf(m, vv[d]); }
        float sum = 0.f;
#pragma unroll
        for (int d = 0; d < 32; ++d) { vv[d] = __expf(vv[d] - m); sum += vv[d]; }
        float inv = QSCALE / sum;
#pragma unroll
        for (int d = 0; d < 32; ++d) qt[p][d] = (_Float16)(vv[d] * inv);
    }
    __syncthreads();

    // B fragments: B[K=d][N=e] = ctx[d][e]  (tiny, L2-hot)
    v16h bfr[2];
#pragma unroll
    for (int eh = 0; eh < 2; ++eh) {
        int e = eh * 16 + (lane & 15);
        int kb = (lane & 16) ? 16 : 0;
        v16h f;
#pragma unroll
        for (int i = 0; i < 16; ++i)
            f[i] = (_Float16)ctx[(unsigned)(bh * 32 + kb + i) * 32 + e];
        bfr[eh] = f;
    }

    for (int tile = wave; tile < 16; tile += 4) {
        int nb = tile * 16;
        v16h af = load_a_frag(&qt[nb + (lane & 15)][0], 0, lane);
#pragma unroll
        for (int eh = 0; eh < 2; ++eh) {
            v8f acc = wmma_f16(af, bfr[eh], zero_v8f());
            unsigned cc = (unsigned)(h * 32 + eh * 16 + (lane & 15));
            unsigned nbase = (unsigned)(n0 + nb + ((lane & 16) ? 8 : 0));
#pragma unroll
            for (int r = 0; r < 8; ++r)
                attn[(nbase + r) * 128 + cc] = (_Float16)acc[r];
        }
    }
}

// ---------------- K6: output projection + bias + rmsnorm ----------------
__global__ void __launch_bounds__(256) k_out(const _Float16* __restrict__ qkv,
                                             const _Float16* __restrict__ Wo,
                                             const float* __restrict__ out_b,
                                             const float* __restrict__ out_norm_g,
                                             float* __restrict__ out) {
    const int b  = blockIdx.y;
    const int n0 = blockIdx.x * 64;
    const int t = threadIdx.x, wave = t >> 5, lane = t & 31;

    __shared__ float as_[128][65];
    __shared__ float psum[4][64];
    __shared__ float sc[64];

    const _Float16* attn = qkv + (unsigned)(b * 384 + 128) * NPIX;  // [n][128]
    const _Float16* wrow = Wo + (unsigned)(wave * 16 + (lane & 15)) * 128;
    v16h afr[4];
#pragma unroll
    for (int ks = 0; ks < 4; ++ks) afr[ks] = load_a_frag(wrow, ks * 32, lane);

    for (int nt = 0; nt < 4; ++nt) {
        v8f acc = zero_v8f();
        const _Float16* arow = attn + (unsigned)(n0 + nt * 16 + (lane & 15)) * 128;
#pragma unroll
        for (int ks = 0; ks < 4; ++ks)
            acc = wmma_f16(afr[ks], load_b_frag(arow, ks * 32, lane), acc);
        int ln = nt * 16 + (lane & 15);
#pragma unroll
        for (int r = 0; r < 8; ++r) {
            int o = wave * 16 + r + ((lane & 16) ? 8 : 0);
            as_[o][ln] = acc[r] + out_b[o];
        }
    }
    __syncthreads();

    const int px = t & 63, qr = t >> 6;
    float part = 0.f;
#pragma unroll
    for (int i = 0; i < 32; ++i) { float v = as_[qr * 32 + i][px]; part += v * v; }
    psum[qr][px] = part;
    __syncthreads();
    if (t < 64) {
        float s = psum[0][t] + psum[1][t] + psum[2][t] + psum[3][t];
        sc[t] = SQRTD / fmaxf(sqrtf(s), 1e-12f);
    }
    __syncthreads();
#pragma unroll
    for (int i = 0; i < 32; ++i) {
        int o = qr * 32 + i;
        out[(unsigned)(b * 128 + o) * NPIX + (unsigned)(n0 + px)] =
            as_[o][px] * sc[px] * out_norm_g[o];
    }
}

// ---------------- launcher ----------------
extern "C" void kernel_launch(void* const* d_in, const int* in_sizes, int n_in,
                              void* d_out, int out_size, void* d_ws, size_t ws_size,
                              hipStream_t stream) {
    const float* x          = (const float*)d_in[0];
    const float* norm_g     = (const float*)d_in[1];
    const float* qkv_w      = (const float*)d_in[2];
    const float* mem_kv     = (const float*)d_in[3];
    const float* out_w      = (const float*)d_in[4];
    const float* out_b      = (const float*)d_in[5];
    const float* out_norm_g = (const float*)d_in[6];
    float* out = (float*)d_out;

    char* ws = (char*)d_ws;
    size_t off = 0;
    _Float16* Wq  = (_Float16*)(ws + off); off += 384 * 128 * sizeof(_Float16);
    _Float16* Wo  = (_Float16*)(ws + off); off += 128 * 128 * sizeof(_Float16);
    float* kmax   = (float*)(ws + off);    off += 2048 * sizeof(float);
    float* ksum   = (float*)(ws + off);    off += 2048 * sizeof(float);
    float* ctx    = (float*)(ws + off);    off += 64 * 32 * 32 * sizeof(float);
    off = (off + 255) & ~(size_t)255;
    _Float16* qkv = (_Float16*)(ws + off); off += (size_t)NB * 384 * NPIX * sizeof(_Float16);

    k_weights<<<192, 256, 0, stream>>>(qkv_w, out_w, Wq, Wo);
    k_rmsnorm_qkv<<<dim3(NPIX / 64, NB), 256, 0, stream>>>(x, norm_g, Wq, qkv);
    k_kstats<<<NB * 128, 256, 0, stream>>>(qkv, mem_kv, kmax, ksum);
    k_context<<<NB * HEADS, 128, 0, stream>>>(qkv, mem_kv, kmax, ksum, ctx);
    k_qattn<<<dim3(NPIX / 256, NB * HEADS), 128, 0, stream>>>(qkv, ctx);
    k_out<<<dim3(NPIX / 64, NB), 256, 0, stream>>>(qkv, Wo, out_b, out_norm_g, out);
}
